// Aggregate_38311108280850
// MI455X (gfx1250) — matrix operand order repaired and checked by reading, not compiled, not run on hardware
//
#include <hip/hip_runtime.h>
#include <cstdint>
#include <cstddef>

typedef __attribute__((ext_vector_type(16))) __bf16 v16bf;
typedef __attribute__((ext_vector_type(8)))  float  v8f;

static __device__ __forceinline__ unsigned short f2bf(float f) {
  __bf16 h = (__bf16)f;
  return __builtin_bit_cast(unsigned short, h);
}
static __device__ __forceinline__ float lrelu(float v) {
  return v >= 0.f ? v : 0.1f * v;
}

// Problem constants
#define BATCH 4
#define HDIM 96
#define WDIM 96
#define NFEAT 64

// ---------------------------------------------------------------------------
// Weight packing: A-fragment order for v_wmma_f32_16x16x32_bf16.
// A-matrix as packed here (M on lanes, K on regs): K = e + 16*half, matched by B
// GEMM-K enumeration (conv): tap-major, channel-minor:
//   K = (kh*KW + kw)*CIN + cin   (CIN multiple of 32 -> each 32-chunk = 1 tap)
// Layout: [kt][mt][lane(32)][e(16)] ushort (bf16 bits). Each lane reads 32B.
// ---------------------------------------------------------------------------
template<int CIN, int KH, int KW, bool DUAL>
__global__ void pack_conv(const float* __restrict__ w,
                          unsigned short* __restrict__ out,
                          int COUT, int MT) {
  constexpr int KTOT = CIN * KH * KW;
  constexpr int KCH  = KTOT / 32;
  const int total = KCH * MT * 512;
  for (int t = blockIdx.x * blockDim.x + threadIdx.x; t < total;
       t += gridDim.x * blockDim.x) {
    int e    = t & 15;
    int lane = (t >> 4) & 31;
    int mt   = (t >> 9) % MT;
    int kt   = t / (512 * MT);
    int m    = mt * 16 + (lane & 15);
    int half = lane >> 4;
    int K    = kt * 32 + e + half * 16;
    float v = 0.f;
    if (m < COUT) {
      int tap = K / CIN, cin = K % CIN;
      int kh = tap / KW, kw = tap % KW;
      constexpr int CSTR = DUAL ? CIN * 2 : CIN;
      v = w[(((size_t)m * CSTR + cin) * KH + kh) * KW + kw];
      if (DUAL)
        v += w[(((size_t)m * CSTR + cin + CIN) * KH + kh) * KW + kw];
    }
    out[t] = f2bf(v);
  }
}

// conv_nn expanded: 9 GEMM weight matrices A_s[64 x 1728] (zero-padded taps).
// K order here matches the gathered-B layout: K = ci*9 + (u*3+v).
__global__ void pack_wnn(const float* __restrict__ wnn,
                         unsigned short* __restrict__ out) {
  const int total = 9 * 54 * 4 * 512;
  for (int t = blockIdx.x * blockDim.x + threadIdx.x; t < total;
       t += gridDim.x * blockDim.x) {
    int e    = t & 15;
    int lane = (t >> 4) & 31;
    int mt   = (t >> 9) & 3;
    int kt   = (t >> 11) % 54;
    int s    = t / 110592;
    int m    = mt * 16 + (lane & 15);
    int K    = kt * 32 + e + (lane >> 4) * 16;
    int ci = K / 9, pos = K % 9;
    int u = pos / 3, v = pos % 3;
    int su = s / 3, sv = s % 3;
    int ku = u - su + 1, kv = v - sv + 1;
    float val = 0.f;
    if (ku >= 0 && ku < 3 && kv >= 0 && kv < 3)
      val = wnn[(((size_t)m * 192 + ci) * 3 + ku) * 3 + kv];
    out[t] = f2bf(val);
  }
}

// ---------------------------------------------------------------------------
// Generic implicit-GEMM conv via WMMA.
// Block = MT*2 waves; each wave owns exactly one (mt, nt) output tile, so the
// accumulator is a single v8f that never leaves VGPRs.
// LDS tile is channel-minor: sx[(kh*XC + col)*CINP + cin]; with the tap-major
// K order a B fragment is 16 consecutive channels -> 2 x ds_load_b128.
// B as loaded here (32x16 bf16, N on lanes, K on regs): K = e + 16*half, same as A.
// ---------------------------------------------------------------------------
template<int CIN, int KH, int KW, int MT, int COUT, bool LEAKY>
__global__ __launch_bounds__(MT * 64)
void conv_wmma(const float* __restrict__ x, const uint4* __restrict__ wfrag,
               const float* __restrict__ bias, float* __restrict__ y) {
  constexpr int THREADS = MT * 64;            // MT*2 waves
  constexpr int PH = (KH - 1) / 2, PW = (KW - 1) / 2;
  constexpr int XC = 31 + KW;                 // staged columns
  constexpr int CINP = CIN + 16;              // padded channel stride (32B mult)
  constexpr int KTOT = CIN * KH * KW;
  constexpr int KCH  = KTOT / 32;
  constexpr int MPAD = MT * 16;

  __shared__ __bf16 sx[KH * XC * CINP];

  const int tid = threadIdx.x;
  const int x0  = blockIdx.x * 32;
  const int y0  = blockIdx.y;
  const int b   = blockIdx.z;

  // Stage halo tile, channel-minor (contiguous LDS writes).
  for (int t = tid; t < KH * XC * CIN; t += THREADS) {
    int pos = t / CIN, cin = t % CIN;
    int kh = pos / XC, cc = pos % XC;
    int row = y0 - PH + kh;
    int col = x0 - PW + cc;
    float v = 0.f;
    if (row >= 0 && row < HDIM && col >= 0 && col < WDIM)
      v = x[((size_t)(b * CIN + cin) * HDIM + row) * WDIM + col];
    sx[pos * CINP + cin] = (__bf16)v;
  }
  __syncthreads();

  const int wid   = tid >> 5;
  const int lane  = tid & 31;
  const int half  = lane >> 4;
  const int nlane = lane & 15;
  const int mt    = wid >> 1;    // one tile per wave
  const int nt    = wid & 1;

  v8f acc;
  #pragma unroll
  for (int r = 0; r < 8; ++r) acc[r] = 0.f;

  const int n = nt * 16 + nlane;
  for (int kt = 0; kt < KCH; ++kt) {
    const int kb   = kt * 32;
    const int tap  = kb / CIN;          // constant across the 32-chunk
    const int kh   = tap / KW, kw = tap % KW;
    const int cin0 = kb % CIN + half * 16;
    const uint4* q =
        (const uint4*)&sx[(kh * XC + n + kw) * CINP + cin0];
    union { uint4 q2[2]; v16bf v; } ub;
    ub.q2[0] = q[0]; ub.q2[1] = q[1];

    const uint4* p = wfrag + (((size_t)kt * MT + mt) * 32 + lane) * 2;
    union { uint4 q2[2]; v16bf v; } ua;
    ua.q2[0] = p[0]; ua.q2[1] = p[1];

    acc = __builtin_amdgcn_wmma_f32_16x16x32_bf16(
        false, ua.v, false, ub.v, (short)0, acc, false, false);
  }

  const int xg = x0 + n;
  #pragma unroll
  for (int r = 0; r < 8; ++r) {
    int m = mt * 16 + half * 8 + r;
    float v = acc[r] + (m < COUT ? bias[m] : 0.f);
    if (LEAKY) v = lrelu(v);
    y[((size_t)(b * MPAD + m) * HDIM + y0) * WDIM + xg] = v;
  }
}

// ---------------------------------------------------------------------------
// Channel-wise L2 normalization, f32 -> bf16.
// ---------------------------------------------------------------------------
__global__ void normalize_k(const float* __restrict__ x,
                            unsigned short* __restrict__ nref) {
  int p = blockIdx.x * blockDim.x + threadIdx.x;
  if (p >= BATCH * HDIM * WDIM) return;
  int b = p / (HDIM * WDIM);
  int yx = p % (HDIM * WDIM);
  float ss = 0.f;
  for (int c = 0; c < NFEAT; ++c) {
    float v = x[(size_t)(b * NFEAT + c) * HDIM * WDIM + yx];
    ss += v * v;
  }
  float inv = 1.f / fmaxf(sqrtf(ss), 1e-12f);
  for (int c = 0; c < NFEAT; ++c) {
    float v = x[(size_t)(b * NFEAT + c) * HDIM * WDIM + yx] * inv;
    nref[(size_t)(b * NFEAT + c) * HDIM * WDIM + yx] = f2bf(v);
  }
}

// ---------------------------------------------------------------------------
// Correlation (patch 7, kernel 3) + top-3 indices per pixel.
// corr[o=(ph,pw)](y,x) = sum_{u,v in 3x3, c} f(y+u-1,x+v-1)*f(y+u-1+ph-3, x+v-1+pw-3)
// 16x16 pixel tile; 24x24 x 32ch bf16 feature tile in LDS, two channel passes.
// All corr[] accesses constant-indexed (unrolled) so the 49 accumulators stay
// in VGPRs (no scratch).
// ---------------------------------------------------------------------------
__global__ __launch_bounds__(256)
void corr_topk(const unsigned short* __restrict__ nref_u,
               int* __restrict__ top3) {
  __shared__ __bf16 f2t[32 * 24 * 24];
  const __bf16* nref = (const __bf16*)nref_u;
  const int tx = threadIdx.x & 15, ty = threadIdx.x >> 4;
  const int bx = blockIdx.x * 16, by = blockIdx.y * 16, b = blockIdx.z;

  float corr[49];
  #pragma unroll
  for (int o = 0; o < 49; ++o) corr[o] = 0.f;

  for (int pass = 0; pass < 2; ++pass) {
    __syncthreads();
    for (int t = threadIdx.x; t < 32 * 24 * 24; t += 256) {
      int c = t / 576, rr = t % 576;
      int row = by + rr / 24 - 4, col = bx + rr % 24 - 4;
      __bf16 v = (__bf16)0.f;
      if (row >= 0 && row < HDIM && col >= 0 && col < WDIM)
        v = nref[((size_t)(b * NFEAT + pass * 32 + c) * HDIM + row) * WDIM + col];
      f2t[t] = v;
    }
    __syncthreads();
    for (int c = 0; c < 32; ++c) {
      const __bf16* base = &f2t[c * 576];
      float f1v[9];
      #pragma unroll
      for (int u = 0; u < 3; ++u)
        #pragma unroll
        for (int v = 0; v < 3; ++v)
          f1v[u * 3 + v] = (float)base[(ty + 3 + u) * 24 + (tx + 3 + v)];
      #pragma unroll
      for (int o = 0; o < 49; ++o) {
        const int dph = o / 7 - 3, dpw = o % 7 - 3;
        float a = 0.f;
        #pragma unroll
        for (int u = 0; u < 3; ++u)
          #pragma unroll
          for (int v = 0; v < 3; ++v)
            a += f1v[u * 3 + v] *
                 (float)base[(ty + 3 + u + dph) * 24 + (tx + 3 + v + dpw)];
        corr[o] += a;
      }
    }
  }

  float v0 = -3.4e38f, v1 = -3.4e38f, v2 = -3.4e38f;
  int i0 = 0, i1 = 0, i2 = 0;
  #pragma unroll
  for (int o = 0; o < 49; ++o) {
    float c = corr[o];
    if (c > v0)      { v2 = v1; i2 = i1; v1 = v0; i1 = i0; v0 = c; i0 = o; }
    else if (c > v1) { v2 = v1; i2 = i1; v1 = c;  i1 = o; }
    else if (c > v2) { v2 = c;  i2 = o; }
  }
  int pidx = ((b * HDIM + (by + ty)) * WDIM + (bx + tx)) * 3;
  top3[pidx] = i0; top3[pidx + 1] = i1; top3[pidx + 2] = i2;
}

// ---------------------------------------------------------------------------
// Fused gather + conv_nn (9 WMMA GEMMs sharing one gathered B) + mask reduce.
// Block = 288 threads = 9 waves: wave w owns output tap s = w and all four
// 16-row M-tiles -> acc[4] fully unrolled, everything stays in VGPRs.
// LDS gathered B = 16 x 1728 bf16, re-used (union) as 64x9x16 f32 afterwards.
// ---------------------------------------------------------------------------
#define AGG_THREADS 288
__global__ __launch_bounds__(AGG_THREADS)
void aggregate_wmma(const float* __restrict__ fea,
                    const uint4* __restrict__ afrag,
                    const float* __restrict__ bnn,
                    const int* __restrict__ top3,
                    const float* __restrict__ masko,
                    float* __restrict__ agg) {
  __shared__ __align__(16) unsigned char smem[16 * 1728 * 2];
  __bf16* gath = (__bf16*)smem;
  float*  gbuf = (float*)smem;
  __shared__ int idxs[48];

  const int tid = threadIdx.x;
  const int x0 = blockIdx.x * 16, y0 = blockIdx.y, b = blockIdx.z;

  if (tid < 48) {
    int n = tid / 3, r = tid % 3;
    idxs[tid] = top3[(((size_t)(b * HDIM + y0) * WDIM) + x0 + n) * 3 + r];
  }
  __syncthreads();

  for (int t = tid; t < 16 * 1728; t += AGG_THREADS) {
    int n = t / 1728, k = t % 1728;
    int ci = k / 9, tap = k % 9;
    int r = ci >> 6, c = ci & 63;
    int o = idxs[n * 3 + r];
    int row = y0 + o / 7 - 4 + tap / 3;
    int col = x0 + n + o % 7 - 4 + tap % 3;
    float v = 0.f;
    if (row >= 0 && row < HDIM && col >= 0 && col < WDIM)
      v = fea[((size_t)(b * NFEAT + c) * HDIM + row) * WDIM + col];
    gath[t] = (__bf16)v;
  }
  __syncthreads();

  const int wid = tid >> 5, lane = tid & 31;
  const int half = lane >> 4, nlane = lane & 15;
  const int s = wid;             // 9 waves <-> 9 output taps

  v8f acc[4];
  #pragma unroll
  for (int i = 0; i < 4; ++i)
    #pragma unroll
    for (int r = 0; r < 8; ++r) acc[i][r] = 0.f;

  for (int kt = 0; kt < 54; ++kt) {
    const uint4* qb = (const uint4*)&gath[nlane * 1728 + kt * 32 + half * 16];
    union { uint4 q2[2]; v16bf v; } ub;
    ub.q2[0] = qb[0]; ub.q2[1] = qb[1];

    const uint4* pbase = afrag + (((size_t)s * 54 + kt) * 4) * 64 + lane * 2;
    #pragma unroll
    for (int i = 0; i < 4; ++i) {
      union { uint4 q2[2]; v16bf v; } ua;
      ua.q2[0] = pbase[i * 64];
      ua.q2[1] = pbase[i * 64 + 1];
      acc[i] = __builtin_amdgcn_wmma_f32_16x16x32_bf16(
          false, ua.v, false, ub.v, (short)0, acc[i], false, false);
    }
  }
  __syncthreads();   // all waves done reading gath; safe to alias as gbuf

  #pragma unroll
  for (int i = 0; i < 4; ++i) {
    #pragma unroll
    for (int r = 0; r < 8; ++r) {
      int m = i * 16 + half * 8 + r;
      gbuf[(m * 9 + s) * 16 + nlane] = lrelu(acc[i][r] + bnn[m]);
    }
  }
  __syncthreads();

  for (int p = tid; p < NFEAT * 16; p += AGG_THREADS) {
    int c = p >> 4, n = p & 15;
    int g = c >> 3;
    float sum = 0.f;
    #pragma unroll
    for (int k = 0; k < 9; ++k)
      sum += gbuf[(c * 9 + k) * 16 + n] *
             masko[((size_t)(b * 80 + g * 9 + k) * HDIM + y0) * WDIM + x0 + n];
    agg[((size_t)(b * NFEAT + c) * HDIM + y0) * WDIM + x0 + n] = lrelu(sum);
  }
}

// concat([ref, agg]) along channels into a 128-channel buffer
__global__ void concat_k(const float* __restrict__ a, const float* __restrict__ g,
                         float* __restrict__ out) {
  const int total = BATCH * 128 * HDIM * WDIM;
  for (int t = blockIdx.x * blockDim.x + threadIdx.x; t < total;
       t += gridDim.x * blockDim.x) {
    int b = t / (128 * HDIM * WDIM);
    int rem = t % (128 * HDIM * WDIM);
    int c = rem / (HDIM * WDIM);
    int yx = rem % (HDIM * WDIM);
    float v = (c < 64)
        ? a[(size_t)(b * 64 + c) * HDIM * WDIM + yx]
        : g[(size_t)(b * 64 + (c - 64)) * HDIM * WDIM + yx];
    out[t] = v;
  }
}

// ---------------------------------------------------------------------------
extern "C" void kernel_launch(void* const* d_in, const int* in_sizes, int n_in,
                              void* d_out, int out_size, void* d_ws, size_t ws_size,
                              hipStream_t stream) {
  (void)in_sizes; (void)n_in; (void)out_size; (void)ws_size;
  const float* ref = (const float*)d_in[0];
  const float* w1  = (const float*)d_in[1];  const float* b1 = (const float*)d_in[2];
  const float* w2  = (const float*)d_in[3];  const float* b2 = (const float*)d_in[4];
  const float* w3  = (const float*)d_in[5];  const float* b3 = (const float*)d_in[6];
  const float* w4  = (const float*)d_in[7];  const float* b4 = (const float*)d_in[8];
  const float* wm  = (const float*)d_in[9];  const float* bm = (const float*)d_in[10];
  const float* wnn = (const float*)d_in[11]; const float* bnn = (const float*)d_in[12];
  const float* wf  = (const float*)d_in[13]; const float* bf = (const float*)d_in[14];
  float* out = (float*)d_out;

  char* base = (char*)d_ws;
  size_t off = 0;
  auto take = [&](size_t bytes) -> char* {
    char* p = base + off;
    off += (bytes + 255) & ~(size_t)255;
    return p;
  };
  const size_t HW = (size_t)HDIM * WDIM;
  unsigned short* nref  = (unsigned short*)take(BATCH * 64 * HW * 2);
  int*            top3  = (int*)take(BATCH * HW * 3 * 4);
  float*          bufA  = (float*)take(BATCH * 64 * HW * 4);
  float*          bufB  = (float*)take(BATCH * 64 * HW * 4);
  float*          masko = (float*)take(BATCH * 80 * HW * 4);
  float*          agg   = (float*)take(BATCH * 64 * HW * 4);
  float*          xcat  = (float*)take(BATCH * 128 * HW * 4);
  unsigned short* fw1   = (unsigned short*)take(18 * 4 * 512 * 2);
  unsigned short* fw2   = (unsigned short*)take(18 * 4 * 512 * 2);
  unsigned short* fw3   = (unsigned short*)take(14 * 4 * 512 * 2);
  unsigned short* fw4   = (unsigned short*)take(14 * 4 * 512 * 2);
  unsigned short* fwm   = (unsigned short*)take(18 * 5 * 512 * 2);
  unsigned short* fwf   = (unsigned short*)take(36 * 4 * 512 * 2);
  unsigned short* fwnn  = (unsigned short*)take((size_t)9 * 54 * 4 * 512 * 2);

  // Weight packing (cheap, deterministic, every call)
  pack_conv<64, 3, 3, true ><<<144, 256, 0, stream>>>(w1, fw1, 64, 4);
  pack_conv<64, 3, 3, false><<<144, 256, 0, stream>>>(w2, fw2, 64, 4);
  pack_conv<64, 7, 1, false><<<112, 256, 0, stream>>>(w3, fw3, 64, 4);
  pack_conv<64, 1, 7, false><<<112, 256, 0, stream>>>(w4, fw4, 64, 4);
  pack_conv<64, 3, 3, false><<<180, 256, 0, stream>>>(wm, fwm, 72, 5);
  pack_conv<128, 3, 3, false><<<288, 256, 0, stream>>>(wf, fwf, 64, 4);
  pack_wnn<<<(9 * 54 * 4 * 512 + 255) / 256, 256, 0, stream>>>(wnn, fwnn);

  // Correlation branch
  normalize_k<<<(BATCH * HW + 255) / 256, 256, 0, stream>>>(ref, nref);
  corr_topk<<<dim3(WDIM / 16, HDIM / 16, BATCH), 256, 0, stream>>>(nref, top3);

  // Mask branch (WMMA convs)
  conv_wmma<64, 3, 3, 4, 64, true ><<<dim3(3, HDIM, BATCH), 256, 0, stream>>>(
      ref, (const uint4*)fw1, b1, bufA);
  conv_wmma<64, 3, 3, 4, 64, true ><<<dim3(3, HDIM, BATCH), 256, 0, stream>>>(
      bufA, (const uint4*)fw2, b2, bufB);
  conv_wmma<64, 7, 1, 4, 64, true ><<<dim3(3, HDIM, BATCH), 256, 0, stream>>>(
      bufB, (const uint4*)fw3, b3, bufA);
  conv_wmma<64, 1, 7, 4, 64, true ><<<dim3(3, HDIM, BATCH), 256, 0, stream>>>(
      bufA, (const uint4*)fw4, b4, bufB);
  conv_wmma<64, 3, 3, 5, 72, false><<<dim3(3, HDIM, BATCH), 320, 0, stream>>>(
      bufB, (const uint4*)fwm, bm, masko);

  // Gather + conv_nn + mask-weighted aggregation
  aggregate_wmma<<<dim3(WDIM / 16, HDIM, BATCH), AGG_THREADS, 0, stream>>>(
      ref, (const uint4*)fwnn, bnn, top3, masko, agg);

  // Final fusion conv
  concat_k<<<(BATCH * 128 * HW + 255) / 256, 256, 0, stream>>>(ref, agg, xcat);
  conv_wmma<128, 3, 3, 4, 64, true><<<dim3(3, HDIM, BATCH), 256, 0, stream>>>(
      xcat, (const uint4*)fwf, bf, out);
}
